// SeqAttentionWithPairBias_44375602102579
// MI455X (gfx1250) — matrix operand order repaired and checked by reading, not compiled, not run on hardware
//
#include <hip/hip_runtime.h>
#include <hip/hip_bf16.h>
#include <hip/hip_fp16.h>

// ---------------------------------------------------------------------------
// SeqAttentionWithPairBias for MI455X (gfx1250, wave32, WMMA bf16)
// N=2, L=512, c_s=384, c_z=128, H=12, d=32
// ---------------------------------------------------------------------------

typedef __attribute__((ext_vector_type(16))) __bf16 bf16x16;
typedef __attribute__((ext_vector_type(8)))  float  floatx8;

#define NB   2
#define LL   512
#define CS   384
#define CZ   128
#define HH   12
#define DD   32
#define ROWS (NB*LL)          // 1024
#define KT_CS (CS/32)         // 12 k-tiles over c_s
#define NT_CS (CS/16)         // 24 n-tiles over c_s
#define MT_R  (ROWS/16)       // 64 m-tiles over rows

static __device__ __forceinline__ float warp_sum(float v) {
    for (int off = 16; off > 0; off >>= 1) v += __shfl_xor(v, off, 32);
    return v;
}

static __device__ __forceinline__ floatx8 wmma_bf16(bf16x16 a, bf16x16 b, floatx8 c) {
    // v_wmma_f32_16x16x32_bf16  D = A(16x32) * B(32x16) + C
    return __builtin_amdgcn_wmma_f32_16x16x32_bf16(false, a, false, b, (short)0, c, false, false);
}

// A tile (16x32 bf16) from row-major [M][K] memory.
// k order used here (contiguous): lanes 0-15: row=lane, elems 0-15 -> K 0..15
//             lanes 16-31: row=lane-16, K 16..31.
static __device__ __forceinline__ bf16x16 load_a_rm(const __bf16* base, int ld, int lane) {
    int half = lane >> 4, m = lane & 15;
    const __bf16* r = base + m * ld + half * 16;
    union { bf16x16 v; uint4 q[2]; } u;
    u.q[0] = *(const uint4*)(r);
    u.q[1] = *(const uint4*)(r + 8);
    return u.v;
}

// B tile (32x16 bf16) from memory laid out [N rows][K contiguous].
// k order used here (contiguous): lanes 0-15: col=lane, elems -> K 0..15; lanes 16-31: K 16..31.
static __device__ __forceinline__ bf16x16 load_b_nk(const __bf16* colbase, int ld, int k0, int lane) {
    const __bf16* r = colbase + (lane & 15) * ld + k0 + ((lane >> 4) << 4);
    union { bf16x16 v; uint4 q[2]; } u;
    u.q[0] = *(const uint4*)(r);
    u.q[1] = *(const uint4*)(r + 8);
    return u.v;
}

// B tile from pre-packed buffer: [tile][lane][8 x u32(bf16 pair)]
static __device__ __forceinline__ bf16x16 load_b_packed(const unsigned* p, int tile, int lane) {
    const uint4* q = (const uint4*)(p + (((size_t)tile) * 32 + lane) * 8);
    union { bf16x16 v; uint4 q[2]; } u;
    u.q[0] = q[0]; u.q[1] = q[1];
    return u.v;
}

// ---------------------------------------------------------------------------
// K0a: pack a 384x384 fp32 weight (row-major [K][N]) into WMMA B-tile layout
// ---------------------------------------------------------------------------
__global__ void k_pack_w(const float* __restrict__ src, unsigned* __restrict__ dst) {
    int t = blockIdx.x * blockDim.x + threadIdx.x;   // KT_CS*NT_CS*32 = 9216
    if (t >= KT_CS * NT_CS * 32) return;
    int lane = t & 31;
    int tile = t >> 5;
    int nt = tile % NT_CS, kt = tile / NT_CS;
    int n = nt * 16 + (lane & 15);
    int kbase = kt * 32 + ((lane >> 4) << 4);
    unsigned* out = dst + ((size_t)tile * 32 + lane) * 8;
#pragma unroll
    for (int u = 0; u < 8; u++) {
        int k0 = kbase + 2 * u;
        union { __bf16 h[2]; unsigned u32; } p;
        p.h[0] = (__bf16)src[(size_t)k0 * CS + n];
        p.h[1] = (__bf16)src[(size_t)(k0 + 1) * CS + n];
        out[u] = p.u32;
    }
}

// ---------------------------------------------------------------------------
// K0b: pack wz[H=12][128] transposed into B-tile layout [K=c 128][N=h 16-pad]
//      4 k-tiles x 1 n-tile
// ---------------------------------------------------------------------------
__global__ void k_pack_wz(const float* __restrict__ wz, unsigned* __restrict__ dst) {
    int t = blockIdx.x * blockDim.x + threadIdx.x;   // 4*32 = 128
    if (t >= 4 * 32) return;
    int lane = t & 31, kt = t >> 5;
    int h = lane & 15;
    int kbase = kt * 32 + ((lane >> 4) << 3);
    unsigned* out = dst + ((size_t)kt * 32 + lane) * 8;
#pragma unroll
    for (int u = 0; u < 8; u++) {
        int c0 = kbase + 2 * u + (u >= 4 ? 8 : 0);
        union { __bf16 hh[2]; unsigned u32; } p;
        float lo = (h < HH) ? wz[h * CZ + c0]     : 0.0f;
        float hi = (h < HH) ? wz[h * CZ + c0 + 1] : 0.0f;
        p.hh[0] = (__bf16)lo; p.hh[1] = (__bf16)hi;
        out[u] = p.u32;
    }
}

// ---------------------------------------------------------------------------
// K1: layernorm(s) -> bf16, one wave per row of 384
// ---------------------------------------------------------------------------
__global__ void k_ln_s(const float* __restrict__ s, const float* __restrict__ w,
                       const float* __restrict__ b, __bf16* __restrict__ sn) {
    int row = blockIdx.x * (blockDim.x >> 5) + (threadIdx.x >> 5);
    int lane = threadIdx.x & 31;
    if (row >= ROWS) return;
    const float* x = s + (size_t)row * CS;
    float vals[12];
    float sum = 0.f;
#pragma unroll
    for (int t = 0; t < 3; t++) {
        float4 v4 = *(const float4*)(x + t * 128 + lane * 4);
        vals[t*4+0] = v4.x; vals[t*4+1] = v4.y; vals[t*4+2] = v4.z; vals[t*4+3] = v4.w;
        sum += v4.x + v4.y + v4.z + v4.w;
    }
    float mean = warp_sum(sum) * (1.0f / CS);
    float vs = 0.f;
#pragma unroll
    for (int i = 0; i < 12; i++) { float d0 = vals[i] - mean; vs += d0 * d0; }
    float inv = rsqrtf(warp_sum(vs) * (1.0f / CS) + 1e-5f);
    __bf16* o = sn + (size_t)row * CS;
#pragma unroll
    for (int t = 0; t < 3; t++)
#pragma unroll
        for (int u = 0; u < 4; u++) {
            int c = t * 128 + lane * 4 + u;
            o[c] = (__bf16)((vals[t*4+u] - mean) * inv * w[c] + b[c]);
        }
}

// ---------------------------------------------------------------------------
// K2: fused q/k/v/g projections: [1024,384] x [384,384] via WMMA bf16
// ---------------------------------------------------------------------------
__global__ void k_proj(const __bf16* __restrict__ sn,
                       const unsigned* __restrict__ wq_p, const unsigned* __restrict__ wk_p,
                       const unsigned* __restrict__ wv_p, const unsigned* __restrict__ wg_p,
                       const float* __restrict__ bg,
                       __bf16* __restrict__ qb, __bf16* __restrict__ kb,
                       __bf16* __restrict__ vtb, float* __restrict__ gbuf) {
    int wid = blockIdx.x * (blockDim.x >> 5) + (threadIdx.x >> 5);
    int lane = threadIdx.x & 31;
    if (wid >= 4 * MT_R * NT_CS) return;
    int which = wid / (MT_R * NT_CS);
    int rem = wid % (MT_R * NT_CS);
    int mt = rem / NT_CS, nt = rem % NT_CS;
    const unsigned* wp = (which == 0) ? wq_p : (which == 1) ? wk_p : (which == 2) ? wv_p : wg_p;
    floatx8 c = {};
#pragma unroll
    for (int kt = 0; kt < KT_CS; kt++) {
        bf16x16 a = load_a_rm(sn + (size_t)mt * 16 * CS + kt * 32, CS, lane);
        bf16x16 b = load_b_packed(wp, kt * NT_CS + nt, lane);
        c = wmma_bf16(a, b, c);
    }
    int half = lane >> 4, col = lane & 15;
    int gcol = nt * 16 + col;
    int h = gcol >> 5, dd = gcol & 31;
#pragma unroll
    for (int v = 0; v < 8; v++) {
        int grow = mt * 16 + v + 8 * half;       // 0..1023
        int n = grow >> 9, l = grow & (LL - 1);
        size_t nh = (size_t)(n * HH + h);
        if (which == 0) {
            qb[(nh * LL + l) * DD + dd] = (__bf16)(c[v] * 0.17677669529663687f);
        } else if (which == 1) {
            kb[(nh * LL + l) * DD + dd] = (__bf16)c[v];
        } else if (which == 2) {
            vtb[(nh * DD + dd) * LL + l] = (__bf16)c[v];
        } else {
            float x = c[v] + bg[gcol];
            gbuf[(size_t)grow * CS + gcol] = 1.0f / (1.0f + __expf(-x));
        }
    }
}

// ---------------------------------------------------------------------------
// K3: layernorm(z) + pair bias einsum, WMMA formulation (HBM-bound: 268 MB)
// One wave handles a 16-row (j) tile of z[n,i,j,:]:
//   - A-layout f32 loads (each row split across lanes l / l+16)
//   - per-row LN stats with a single shfl_xor(16)
//   - normalize -> bf16 A tile, 4x WMMA vs packed wz^T  -> bias[n,h,i,j]
// ---------------------------------------------------------------------------
__global__ void k_bias(const float* __restrict__ z, const float* __restrict__ w,
                       const float* __restrict__ b, const unsigned* __restrict__ wz_p,
                       float* __restrict__ bias) {
    int wid = blockIdx.x * (blockDim.x >> 5) + (threadIdx.x >> 5);
    int lane = threadIdx.x & 31;
    if (wid >= NB * LL * (LL / 16)) return;       // 32768 tiles
    int jt = wid & 31;
    int ni = wid >> 5;                            // n*512 + i
    int n = ni >> 9, i = ni & (LL - 1);
    int j0 = jt * 16;
    int half = lane >> 4, m = lane & 15;
    const float* rowp = z + ((size_t)ni * LL + j0 + m) * CZ;
    __builtin_prefetch(rowp + 16 * CZ, 0, 0);     // next j-tile of this (n,i)

    // gather this lane's 64 elements in A-layout order (f32)
    float vals[64];
    float sum = 0.f, sumsq = 0.f;
#pragma unroll
    for (int t = 0; t < 4; t++) {
        int cA = t * 32 + half * 8;
        float4 a0 = *(const float4*)(rowp + cA);
        float4 a1 = *(const float4*)(rowp + cA + 4);
        float4 b0 = *(const float4*)(rowp + cA + 16);
        float4 b1 = *(const float4*)(rowp + cA + 20);
        float* vt = &vals[t * 16];
        vt[0]=a0.x; vt[1]=a0.y; vt[2]=a0.z; vt[3]=a0.w;
        vt[4]=a1.x; vt[5]=a1.y; vt[6]=a1.z; vt[7]=a1.w;
        vt[8]=b0.x; vt[9]=b0.y; vt[10]=b0.z; vt[11]=b0.w;
        vt[12]=b1.x; vt[13]=b1.y; vt[14]=b1.z; vt[15]=b1.w;
#pragma unroll
        for (int e = 0; e < 16; e++) { sum += vt[e]; sumsq += vt[e]*vt[e]; }
    }
    // row stats: other half of row lives in lane^16
    sum   += __shfl_xor(sum, 16, 32);
    sumsq += __shfl_xor(sumsq, 16, 32);
    float mean = sum * (1.0f / CZ);
    float var  = sumsq * (1.0f / CZ) - mean * mean;
    float inv  = rsqrtf(var + 1e-5f);

    floatx8 c = {};
#pragma unroll
    for (int t = 0; t < 4; t++) {
        int cA = t * 32 + half * 8;
        float4 w0 = *(const float4*)(w + cA);
        float4 w1 = *(const float4*)(w + cA + 4);
        float4 w2 = *(const float4*)(w + cA + 16);
        float4 w3 = *(const float4*)(w + cA + 20);
        float4 bb0 = *(const float4*)(b + cA);
        float4 bb1 = *(const float4*)(b + cA + 4);
        float4 bb2 = *(const float4*)(b + cA + 16);
        float4 bb3 = *(const float4*)(b + cA + 20);
        float wv[16] = {w0.x,w0.y,w0.z,w0.w, w1.x,w1.y,w1.z,w1.w,
                        w2.x,w2.y,w2.z,w2.w, w3.x,w3.y,w3.z,w3.w};
        float bv[16] = {bb0.x,bb0.y,bb0.z,bb0.w, bb1.x,bb1.y,bb1.z,bb1.w,
                        bb2.x,bb2.y,bb2.z,bb2.w, bb3.x,bb3.y,bb3.z,bb3.w};
        union { bf16x16 v; __bf16 e[16]; } a;
#pragma unroll
        for (int e = 0; e < 16; e++) {
            float zn = (vals[t * 16 + e] - mean) * inv * wv[e] + bv[e];
            a.e[e] = (__bf16)zn;
        }
        bf16x16 bt = load_b_packed(wz_p, t, lane);
        c = wmma_bf16(a.v, bt, c);
    }
    // C: rows = j offset (v + 8*half), cols = head
    int hcol = lane & 15;
    if (hcol < HH) {
        float* op = bias + (((size_t)(n * HH + hcol) * LL + i) * LL) + j0 + half * 8;
        float4 o0 = {c[0], c[1], c[2], c[3]};
        float4 o1 = {c[4], c[5], c[6], c[7]};
        *(float4*)(op)     = o0;
        *(float4*)(op + 4) = o1;
    }
}

// ---------------------------------------------------------------------------
// K4: attention per (n,h,16-row i-tile), one wave per block.
// Pass1: 32x WMMA QK^T + bias -> LDS (f16, padded stride 520), track row max.
// Pass2: rebuild P (bf16, A-layout) from LDS with expf, 2x WMMA vs V^T.
// Epilogue: /rowsum, * gate, write bf16 row-major [1024,384].
// ---------------------------------------------------------------------------
#define SSTR 520   // f16 row stride: 520 halves = 260 words; m*260 %64 = 4m -> distinct banks

__global__ __launch_bounds__(32) void k_attn(const __bf16* __restrict__ qb,
                                             const __bf16* __restrict__ kb,
                                             const __bf16* __restrict__ vtb,
                                             const float* __restrict__ bias,
                                             const float* __restrict__ gbuf,
                                             __bf16* __restrict__ og) {
    __shared__ __half S[16 * SSTR];    // ~16.3 KB -> high workgroup residency per WGP
    __shared__ float rmax_s[16];
    __shared__ float rsum_s[16];
    int lane = threadIdx.x;
    int bid = blockIdx.x;                 // (n*12+h)*32 + itile
    int itile = bid & 31;
    int nh = bid >> 5;
    int n = nh / HH, h = nh % HH;
    int ibase = itile * 16;
    const __bf16* qptr = qb  + ((size_t)nh * LL + ibase) * DD;
    const __bf16* kcol = kb  + (size_t)nh * LL * DD;
    const __bf16* vcol = vtb + (size_t)nh * DD * LL;
    const float*  bptr = bias + ((size_t)nh * LL + ibase) * LL;

    bf16x16 aq = load_a_rm(qptr, DD, lane);
    int half = lane >> 4, col = lane & 15;

    float rmax[8];
#pragma unroll
    for (int v = 0; v < 8; v++) rmax[v] = -1e30f;

    // ---- pass 1: logits tiles (N=16 wide), d=32 = one WMMA each ----
    for (int jt = 0; jt < LL / 16; jt++) {
        int j0 = jt * 16;
        bf16x16 bk = load_b_nk(kcol + (size_t)j0 * DD, DD, 0, lane);
        floatx8 c = {};
        c = wmma_bf16(aq, bk, c);
#pragma unroll
        for (int v = 0; v < 8; v++) {
            int m = v + 8 * half;
            float sv = c[v] + bptr[(size_t)m * LL + j0 + col];
            S[m * SSTR + j0 + col] = __float2half(sv);
            rmax[v] = fmaxf(rmax[v], sv);
        }
    }
#pragma unroll
    for (int off = 8; off > 0; off >>= 1)
#pragma unroll
        for (int v = 0; v < 8; v++)
            rmax[v] = fmaxf(rmax[v], __shfl_xor(rmax[v], off, 32));
    if (lane == 0)  for (int v = 0; v < 8; v++) rmax_s[v] = rmax[v];
    if (lane == 16) for (int v = 0; v < 8; v++) rmax_s[8 + v] = rmax[v];
    __syncthreads();

    // ---- pass 2: P = exp(S-max) in A layout, O += P @ V^T ----
    int m = col;                       // A-layout row for this lane
    float rm = rmax_s[m];
    float rsum = 0.0f;
    floatx8 o0 = {}, o1 = {};
    for (int jt = 0; jt < LL / 32; jt++) {
        int j0 = jt * 32;
        const __half* srow = &S[m * SSTR + j0 + half * 16];
        union { uint4 q; __half2 h2[4]; } u0, u1;
        u0.q = *(const uint4*)(srow);        // elems 0..7
        u1.q = *(const uint4*)(srow + 8);    // elems 8..15
        float vals[16];
#pragma unroll
        for (int t = 0; t < 4; t++) {
            float2 f0 = __half22float2(u0.h2[t]);
            float2 f1 = __half22float2(u1.h2[t]);
            vals[2*t]     = f0.x; vals[2*t+1]     = f0.y;
            vals[8+2*t]   = f1.x; vals[8+2*t+1]   = f1.y;
        }
        union { bf16x16 v; __bf16 e[16]; } p;
#pragma unroll
        for (int e = 0; e < 16; e++) {
            float pe = __expf(vals[e] - rm);
            rsum += pe;
            p.e[e] = (__bf16)pe;
        }
        bf16x16 b0 = load_b_nk(vcol,           LL, j0, lane);   // dd 0..15
        bf16x16 b1 = load_b_nk(vcol + 16 * LL, LL, j0, lane);   // dd 16..31
        o0 = wmma_bf16(p.v, b0, o0);
        o1 = wmma_bf16(p.v, b1, o1);
    }
    rsum += __shfl_xor(rsum, 16, 32);
    if (lane < 16) rsum_s[lane] = rsum;
    __syncthreads();

    // ---- epilogue: normalize, gate, store bf16 ----
#pragma unroll
    for (int v = 0; v < 8; v++) {
        int mr = v + 8 * half;
        float inv = 1.0f / rsum_s[mr];
        int row = n * LL + ibase + mr;
        int gcol0 = h * DD + col;
        int gcol1 = h * DD + 16 + col;
        float g0 = gbuf[(size_t)row * CS + gcol0];
        float g1 = gbuf[(size_t)row * CS + gcol1];
        og[(size_t)row * CS + gcol0] = (__bf16)(o0[v] * inv * g0);
        og[(size_t)row * CS + gcol1] = (__bf16)(o1[v] * inv * g1);
    }
}

// ---------------------------------------------------------------------------
// K5: output projection: out = og[1024,384] x wo[384,384] + bo  (WMMA bf16)
// ---------------------------------------------------------------------------
__global__ void k_outproj(const __bf16* __restrict__ og, const unsigned* __restrict__ wo_p,
                          const float* __restrict__ bo, float* __restrict__ out) {
    int wid = blockIdx.x * (blockDim.x >> 5) + (threadIdx.x >> 5);
    int lane = threadIdx.x & 31;
    if (wid >= MT_R * NT_CS) return;
    int mt = wid / NT_CS, nt = wid % NT_CS;
    floatx8 c = {};
#pragma unroll
    for (int kt = 0; kt < KT_CS; kt++) {
        bf16x16 a = load_a_rm(og + (size_t)mt * 16 * CS + kt * 32, CS, lane);
        bf16x16 b = load_b_packed(wo_p, kt * NT_CS + nt, lane);
        c = wmma_bf16(a, b, c);
    }
    int half = lane >> 4, col = lane & 15;
    int gcol = nt * 16 + col;
    float bv = bo[gcol];
#pragma unroll
    for (int v = 0; v < 8; v++) {
        int grow = mt * 16 + v + 8 * half;
        out[(size_t)grow * CS + gcol] = c[v] + bv;
    }
}

// ---------------------------------------------------------------------------
extern "C" void kernel_launch(void* const* d_in, const int* in_sizes, int n_in,
                              void* d_out, int out_size, void* d_ws, size_t ws_size,
                              hipStream_t stream) {
    const float* s      = (const float*)d_in[0];
    const float* z      = (const float*)d_in[1];
    const float* ln_s_w = (const float*)d_in[2];
    const float* ln_s_b = (const float*)d_in[3];
    const float* ln_z_w = (const float*)d_in[4];
    const float* ln_z_b = (const float*)d_in[5];
    const float* wz     = (const float*)d_in[6];
    const float* wq     = (const float*)d_in[7];
    const float* wk     = (const float*)d_in[8];
    const float* wv     = (const float*)d_in[9];
    const float* wg     = (const float*)d_in[10];
    const float* bg     = (const float*)d_in[11];
    const float* wo     = (const float*)d_in[12];
    const float* bo     = (const float*)d_in[13];
    float* out = (float*)d_out;

    // workspace carve (256B aligned)
    char* ws = (char*)d_ws;
    size_t off = 0;
    auto carve = [&](size_t bytes) {
        void* p = ws + off;
        off = (off + bytes + 255) & ~(size_t)255;
        return p;
    };
    __bf16*   sn   = (__bf16*)  carve((size_t)ROWS * CS * 2);
    unsigned* wq_p = (unsigned*)carve((size_t)CS * CS * 2);
    unsigned* wk_p = (unsigned*)carve((size_t)CS * CS * 2);
    unsigned* wv_p = (unsigned*)carve((size_t)CS * CS * 2);
    unsigned* wg_p = (unsigned*)carve((size_t)CS * CS * 2);
    unsigned* wo_p = (unsigned*)carve((size_t)CS * CS * 2);
    unsigned* wz_p = (unsigned*)carve((size_t)4 * 32 * 8 * 4);   // 4 B-tiles
    __bf16*   qb   = (__bf16*)  carve((size_t)NB * HH * LL * DD * 2);
    __bf16*   kb   = (__bf16*)  carve((size_t)NB * HH * LL * DD * 2);
    __bf16*   vtb  = (__bf16*)  carve((size_t)NB * HH * DD * LL * 2);
    float*    gbuf = (float*)   carve((size_t)ROWS * CS * 4);
    float*    biasb= (float*)   carve((size_t)NB * HH * LL * LL * 4);
    __bf16*   og   = (__bf16*)  carve((size_t)ROWS * CS * 2);

    // K0: pack weights into WMMA B-tile layout
    {
        int thr = KT_CS * NT_CS * 32;             // 9216
        int blk = (thr + 255) / 256;
        k_pack_w<<<blk, 256, 0, stream>>>(wq, wq_p);
        k_pack_w<<<blk, 256, 0, stream>>>(wk, wk_p);
        k_pack_w<<<blk, 256, 0, stream>>>(wv, wv_p);
        k_pack_w<<<blk, 256, 0, stream>>>(wg, wg_p);
        k_pack_w<<<blk, 256, 0, stream>>>(wo, wo_p);
        k_pack_wz<<<1, 128, 0, stream>>>(wz, wz_p);
    }
    // K1: layernorm(s) -> bf16
    k_ln_s<<<ROWS / 8, 256, 0, stream>>>(s, ln_s_w, ln_s_b, sn);
    // K2: q/k/v/g projections (WMMA)
    k_proj<<<(4 * MT_R * NT_CS) / 8, 256, 0, stream>>>(sn, wq_p, wk_p, wv_p, wg_p,
                                                       bg, qb, kb, vtb, gbuf);
    // K3: layernorm(z) + pair bias via WMMA (streams 268 MB — HBM-bound phase)
    k_bias<<<(NB * LL * (LL / 16)) / 8, 256, 0, stream>>>(z, ln_z_w, ln_z_b, wz_p, biasb);
    // K4: attention (WMMA, softmax through f16 LDS)
    k_attn<<<NB * HH * (LL / 16), 32, 0, stream>>>(qb, kb, vtb, biasb, gbuf, og);
    // K5: output projection (WMMA)
    k_outproj<<<(MT_R * NT_CS) / 8, 256, 0, stream>>>(og, wo_p, bo, out);
}
